// SimpleGAT_79731772883558
// MI455X (gfx1250) — compile-verified, matrix operand order checked by reading, not run on hardware
//
#include <hip/hip_runtime.h>
#include <hip/hip_bf16.h>
#include <math.h>

#define N_NODES   100000
#define N_EDGES   3200000
#define F_INF     128
#define NHEADS    4
#define CHEAD     32
#define HC        128
#define NGROUPS   64
#define NCLS      10
#define NEG_SLOPE 0.2f

typedef __attribute__((ext_vector_type(16))) _Float16 v16h;
typedef __attribute__((ext_vector_type(8)))  float    v8f;

// ---------------------------------------------------------------- utilities
__device__ __forceinline__ unsigned fkey(float f) {
    unsigned u = __float_as_uint(f);
    return (u & 0x80000000u) ? ~u : (u | 0x80000000u);   // order-preserving
}
__device__ __forceinline__ float funkey(unsigned k) {
    unsigned u = (k & 0x80000000u) ? (k ^ 0x80000000u) : ~k;
    return __uint_as_float(u);
}

__global__ void k_zero_u32(unsigned* p, long n) {
    long i = (long)blockIdx.x * blockDim.x + threadIdx.x;
    if (i < n) p[i] = 0u;
}

// ------------------------------------------------- fp32 -> fp16 conversions
__global__ void k_convert_x(const float* __restrict__ x, _Float16* __restrict__ xh) {
    long i = (long)blockIdx.x * blockDim.x + threadIdx.x;
    if (i < (long)N_NODES * F_INF) xh[i] = (_Float16)x[i];
}
// W [K=128, Ncol=128] row-major  ->  Wt [col*128 + k] (column-major) in fp16
__global__ void k_convert_w(const float* __restrict__ W, _Float16* __restrict__ Wt) {
    int i = blockIdx.x * blockDim.x + threadIdx.x;
    if (i < F_INF * HC) {
        int k = i / HC, col = i % HC;
        Wt[col * F_INF + k] = (_Float16)W[i];
    }
}

// -------------------------------------------------------- WMMA GEMM  h = x@W
// 256 threads = 8 waves; wave handles one 16-row tile x all 128 columns.
__global__ void k_gemm_wmma(const _Float16* __restrict__ xh,
                            const _Float16* __restrict__ Wt,
                            float* __restrict__ h) {
    __shared__ _Float16 lw[F_INF * HC];          // 32 KB, W column-major
    int tid = threadIdx.x;
    {   // cooperative preload of Wt into LDS (128-bit chunks)
        const uint4* srcp = (const uint4*)Wt;
        uint4* dstp = (uint4*)lw;
        for (int i = tid; i < F_INF * HC / 8; i += 256) dstp[i] = srcp[i];
    }
    __syncthreads();

    int wave = tid >> 5;
    int lane = tid & 31;
    int rowTile = blockIdx.x * 8 + wave;
    if (rowTile >= N_NODES / 16) return;          // wave-uniform, EXEC stays full
    int row0 = rowTile * 16;
    int m    = lane & 15;
    int kgrp = (lane >> 4) << 3;                  // 0 or 8 (ISA 16-bit A layout)

    v8f acc[8] = {};

    for (int kc = 0; kc < 4; ++kc) {              // K = 4 x 32
        int kbase = kc * 32;
        union { unsigned u[8]; v16h v; } A;
        const unsigned* xrow = (const unsigned*)(xh + (size_t)(row0 + m) * F_INF);
#pragma unroll
        for (int r = 0; r < 8; ++r) {
            int k = kbase + ((r < 4) ? r * 2 : 16 + (r - 4) * 2) + kgrp;
            A.u[r] = xrow[k >> 1];                // pair (k,k+1) contiguous
        }
#pragma unroll
        for (int ct = 0; ct < 8; ++ct) {
            union { unsigned u[8]; v16h v; } B;
            int col = ct * 16 + m;
            const unsigned* wcol = (const unsigned*)(lw + col * F_INF);
#pragma unroll
            for (int r = 0; r < 8; ++r) {
                int k = kbase + ((r < 4) ? r * 2 : 16 + (r - 4) * 2) + kgrp;
                B.u[r] = wcol[k >> 1];
            }
            acc[ct] = __builtin_amdgcn_wmma_f32_16x16x32_f16(
                false, A.v, false, B.v, (short)0, acc[ct], false, false);
        }
    }
    // D layout: lanes 0-15 -> rows j, lanes 16-31 -> rows j+8; col = ct*16 + m
    int rOff = (lane >> 4) << 3;
#pragma unroll
    for (int ct = 0; ct < 8; ++ct)
#pragma unroll
        for (int j = 0; j < 8; ++j)
            h[(size_t)(row0 + rOff + j) * HC + ct * 16 + m] = acc[ct][j];
}

// -------------------------------------------- per-node attention coefficients
__global__ void k_attn_scores(const float* __restrict__ h,
                              const float* __restrict__ att_src,
                              const float* __restrict__ att_dst,
                              float* __restrict__ a_src,
                              float* __restrict__ a_dst) {
    int wave = threadIdx.x >> 5, lane = threadIdx.x & 31;
    int n = blockIdx.x * 8 + wave;
    if (n >= N_NODES) return;
    const float* hr = h + (size_t)n * HC;
    float s[NHEADS], d[NHEADS];
#pragma unroll
    for (int hh = 0; hh < NHEADS; ++hh) {
        float v = hr[hh * CHEAD + lane];
        s[hh] = v * att_src[hh * CHEAD + lane];
        d[hh] = v * att_dst[hh * CHEAD + lane];
    }
#pragma unroll
    for (int off = 16; off > 0; off >>= 1)
#pragma unroll
        for (int hh = 0; hh < NHEADS; ++hh) {
            s[hh] += __shfl_xor(s[hh], off, 32);
            d[hh] += __shfl_xor(d[hh], off, 32);
        }
    if (lane == 0)
#pragma unroll
        for (int hh = 0; hh < NHEADS; ++hh) {
            a_src[(size_t)n * NHEADS + hh] = s[hh];
            a_dst[(size_t)n * NHEADS + hh] = d[hh];
        }
}

// ------------------------------------------------ edge pass 1: segment max
__global__ void k_edge_max(const long long* __restrict__ ei,
                           const float* __restrict__ a_src,
                           const float* __restrict__ a_dst,
                           unsigned* __restrict__ mkey) {
    long e = (long)blockIdx.x * blockDim.x + threadIdx.x;
    if (e >= N_EDGES) return;
    long s = (long)ei[e], d = (long)ei[(long)N_EDGES + e];
#pragma unroll
    for (int hh = 0; hh < NHEADS; ++hh) {
        float le = a_src[s * NHEADS + hh] + a_dst[d * NHEADS + hh];
        le = (le > 0.f) ? le : NEG_SLOPE * le;
        atomicMax(&mkey[d * NHEADS + hh], fkey(le));
    }
}

// ------------------------------------------------ edge pass 2: softmax denom
__global__ void k_edge_denom(const long long* __restrict__ ei,
                             const float* __restrict__ a_src,
                             const float* __restrict__ a_dst,
                             const unsigned* __restrict__ mkey,
                             float* __restrict__ denom) {
    long e = (long)blockIdx.x * blockDim.x + threadIdx.x;
    if (e >= N_EDGES) return;
    long s = (long)ei[e], d = (long)ei[(long)N_EDGES + e];
#pragma unroll
    for (int hh = 0; hh < NHEADS; ++hh) {
        float le = a_src[s * NHEADS + hh] + a_dst[d * NHEADS + hh];
        le = (le > 0.f) ? le : NEG_SLOPE * le;
        float m = funkey(mkey[d * NHEADS + hh]);
        atomicAdd(&denom[d * NHEADS + hh], expf(le - m));
    }
}

// --------------------------------------- edge pass 3: weighted aggregation
// One wave per edge: lane l handles features [4l, 4l+4), head = l>>3.
__global__ void k_edge_aggregate(const long long* __restrict__ ei,
                                 const float* __restrict__ a_src,
                                 const float* __restrict__ a_dst,
                                 const unsigned* __restrict__ mkey,
                                 const float* __restrict__ denom,
                                 const float* __restrict__ h,
                                 float* __restrict__ outb) {
    int lane = threadIdx.x & 31;
    long e = (long)blockIdx.x * 8 + (threadIdx.x >> 5);
    if (e >= N_EDGES) return;
    long s = (long)ei[e], d = (long)ei[(long)N_EDGES + e];
    int hh = lane >> 3;
    float le = a_src[s * NHEADS + hh] + a_dst[d * NHEADS + hh];
    le = (le > 0.f) ? le : NEG_SLOPE * le;
    float m = funkey(mkey[d * NHEADS + hh]);
    float alpha = expf(le - m) / (denom[d * NHEADS + hh] + 1e-16f);
    float4 hv = *(const float4*)(h + (size_t)s * HC + lane * 4);
    float* op = outb + (size_t)d * HC + lane * 4;
    atomicAdd(op + 0, hv.x * alpha);
    atomicAdd(op + 1, hv.y * alpha);
    atomicAdd(op + 2, hv.z * alpha);
    atomicAdd(op + 3, hv.w * alpha);
}

// -------------------------------------- ELU + bias + mean-pool accumulation
// batch is sorted: accumulate runs in registers, flush on group change.
#define NODES_PER_BLOCK 256
__global__ void k_elu_pool(const float* __restrict__ outb,
                           const float* __restrict__ bias,
                           const long long* __restrict__ batch,
                           float* __restrict__ pooled,
                           float* __restrict__ counts) {
    __shared__ int gb[NODES_PER_BLOCK];
    int t = threadIdx.x;                          // 128 threads = features
    int n0 = blockIdx.x * NODES_PER_BLOCK;
    int nEnd = n0 + NODES_PER_BLOCK; if (nEnd > N_NODES) nEnd = N_NODES;
    int cnt = nEnd - n0;
    for (int i = t; i < cnt; i += HC) gb[i] = (int)batch[n0 + i];
    __syncthreads();
    if (cnt <= 0) return;
    float b = bias[t];
    float acc = 0.f; int curg = gb[0]; int lc = 0;
    for (int i = 0; i < cnt; ++i) {
        int g = gb[i];
        if (g != curg) {
            atomicAdd(&pooled[(size_t)curg * HC + t], acc);
            if (t == 0) atomicAdd(&counts[curg], (float)lc);
            acc = 0.f; lc = 0; curg = g;
        }
        float v = outb[(size_t)(n0 + i) * HC + t] + b;
        acc += (v > 0.f) ? v : (expf(v) - 1.f);   // ELU
        ++lc;
    }
    atomicAdd(&pooled[(size_t)curg * HC + t], acc);
    if (t == 0) atomicAdd(&counts[curg], (float)lc);
}

// ------------------------------------------- classifier + log_softmax (tiny)
__global__ void k_classify(const float* __restrict__ pooled,
                           const float* __restrict__ counts,
                           const float* __restrict__ lin_w,
                           const float* __restrict__ lin_b,
                           float* __restrict__ out) {
    int g = threadIdx.x;
    if (g >= NGROUPS) return;
    float inv = 1.f / fmaxf(counts[g], 1.f);
    float logits[NCLS];
#pragma unroll
    for (int c = 0; c < NCLS; ++c) logits[c] = lin_b[c];
    for (int f = 0; f < HC; ++f) {
        float p = pooled[(size_t)g * HC + f] * inv;
#pragma unroll
        for (int c = 0; c < NCLS; ++c) logits[c] += p * lin_w[f * NCLS + c];
    }
    float mx = logits[0];
#pragma unroll
    for (int c = 1; c < NCLS; ++c) mx = fmaxf(mx, logits[c]);
    float sum = 0.f;
#pragma unroll
    for (int c = 0; c < NCLS; ++c) sum += expf(logits[c] - mx);
    float lse = mx + logf(sum);
#pragma unroll
    for (int c = 0; c < NCLS; ++c) out[g * NCLS + c] = logits[c] - lse;
}

// --------------------------------------------------------------- launcher
extern "C" void kernel_launch(void* const* d_in, const int* in_sizes, int n_in,
                              void* d_out, int out_size, void* d_ws, size_t ws_size,
                              hipStream_t stream) {
    const float*     x       = (const float*)d_in[0];
    const long long* ei      = (const long long*)d_in[1];
    const long long* batch   = (const long long*)d_in[2];
    const float*     W       = (const float*)d_in[3];
    const float*     att_src = (const float*)d_in[4];
    const float*     att_dst = (const float*)d_in[5];
    const float*     cbias   = (const float*)d_in[6];
    const float*     lin_w   = (const float*)d_in[7];
    const float*     lin_b   = (const float*)d_in[8];
    float* out = (float*)d_out;

    char* ws = (char*)d_ws;
    size_t off = 0;
    auto alloc = [&](size_t bytes) {
        size_t o = off;
        off = (off + bytes + 255) & ~(size_t)255;
        return o;
    };
    size_t o_xh   = alloc((size_t)N_NODES * F_INF * 2);   // x fp16
    size_t o_wt   = alloc((size_t)F_INF * HC * 2);        // W^T fp16
    size_t o_h    = alloc((size_t)N_NODES * HC * 4);      // h = xW
    size_t o_asrc = alloc((size_t)N_NODES * NHEADS * 4);
    size_t o_adst = alloc((size_t)N_NODES * NHEADS * 4);
    size_t o_mkey = alloc((size_t)N_NODES * NHEADS * 4);  // zeroed region start
    size_t o_den  = alloc((size_t)N_NODES * NHEADS * 4);
    size_t o_out  = alloc((size_t)N_NODES * HC * 4);
    size_t o_pool = alloc((size_t)NGROUPS * HC * 4);
    size_t o_cnt  = alloc((size_t)NGROUPS * 4);
    (void)o_den; (void)ws_size; (void)in_sizes; (void)n_in; (void)out_size;

    _Float16* xh    = (_Float16*)(ws + o_xh);
    _Float16* Wt    = (_Float16*)(ws + o_wt);
    float*    hbuf  = (float*)(ws + o_h);
    float*    asrc  = (float*)(ws + o_asrc);
    float*    adst  = (float*)(ws + o_adst);
    unsigned* mkey  = (unsigned*)(ws + o_mkey);
    float*    denom = (float*)(ws + o_den);
    float*    outb  = (float*)(ws + o_out);
    float*    pool  = (float*)(ws + o_pool);
    float*    cntp  = (float*)(ws + o_cnt);

    // zero all accumulator buffers (mkey..counts are contiguous in the layout)
    long nz = (long)((off - o_mkey) / 4);
    k_zero_u32<<<(nz + 255) / 256, 256, 0, stream>>>(mkey, nz);

    long nx = (long)N_NODES * F_INF;
    k_convert_x<<<(nx + 255) / 256, 256, 0, stream>>>(x, xh);
    k_convert_w<<<(F_INF * HC + 255) / 256, 256, 0, stream>>>(W, Wt);

    int rowTiles = N_NODES / 16;                      // 6250
    k_gemm_wmma<<<(rowTiles + 7) / 8, 256, 0, stream>>>(xh, Wt, hbuf);

    k_attn_scores<<<(N_NODES + 7) / 8, 256, 0, stream>>>(hbuf, att_src, att_dst, asrc, adst);

    int eb = (N_EDGES + 255) / 256;
    k_edge_max  <<<eb, 256, 0, stream>>>(ei, asrc, adst, mkey);
    k_edge_denom<<<eb, 256, 0, stream>>>(ei, asrc, adst, mkey, denom);
    k_edge_aggregate<<<(N_EDGES + 7) / 8, 256, 0, stream>>>(ei, asrc, adst, mkey, denom, hbuf, outb);

    k_elu_pool<<<(N_NODES + NODES_PER_BLOCK - 1) / NODES_PER_BLOCK, HC, 0, stream>>>(
        outb, cbias, batch, pool, cntp);

    k_classify<<<1, NGROUPS, 0, stream>>>(pool, cntp, lin_w, lin_b, out);
}
